// Co_attention_head_45449343926497
// MI455X (gfx1250) — matrix operand order repaired and checked by reading, not compiled, not run on hardware
//
#include <hip/hip_runtime.h>

#define H_DIM   4096
#define N_NODES 1024
#define D_HEAD  1024
#define N_HEADS 4
#define LN_EPS  1e-5f
#define LDS_STRIDE 34   // 32 + 2 pad (ushorts) = 17 DWORDs row pitch

typedef __attribute__((ext_vector_type(16))) __bf16 v16bf;
typedef __attribute__((ext_vector_type(8)))  float  v8f;
typedef __attribute__((ext_vector_type(4)))  unsigned int v4u;
typedef __attribute__((ext_vector_type(8)))  int    v8i;
typedef __attribute__((ext_vector_type(4)))  int    v4i;

#if defined(__AMDGCN__) && __has_builtin(__builtin_amdgcn_tensor_load_to_lds) && \
    __has_builtin(__builtin_amdgcn_s_wait_tensorcnt)
#define USE_TDM 1
#else
#define USE_TDM 0
#endif

// ---- helpers ---------------------------------------------------------------

__device__ __forceinline__ unsigned short f2bf(float f) {
  unsigned int u = __float_as_uint(f);
  u += 0x7FFFu + ((u >> 16) & 1u);   // round-to-nearest-even
  return (unsigned short)(u >> 16);
}

// A fragment, 16x32 bf16 (M x K), ISA 7.12.2 layout.
__device__ __forceinline__ v16bf load_frag_a(const unsigned short* tile, int rowBase, int lane) {
  const int l  = lane & 15;
  const int hi = lane >> 4;
  const unsigned int* p = (const unsigned int*)(tile + (rowBase + l) * LDS_STRIDE);
  union { v16bf v; unsigned int u[8]; } f;
#pragma unroll
  for (int i = 0; i < 4; ++i) f.u[i]     = p[hi * 4 + i];
#pragma unroll
  for (int i = 0; i < 4; ++i) f.u[4 + i] = p[8 + hi * 4 + i];
  return f.v;
}

// B fragment, 32x16 bf16 (K x N); tileT stored K-contiguous per column: [n][k].
__device__ __forceinline__ v16bf load_frag_b(const unsigned short* tileT, int colBase, int lane) {
  const int n  = lane & 15;
  const int hi = lane >> 4;
  const unsigned int* p = (const unsigned int*)(tileT + (colBase + n) * LDS_STRIDE);
  union { v16bf v; unsigned int u[8]; } f;
#pragma unroll
  for (int i = 0; i < 4; ++i) f.u[i]     = p[hi * 4 + i];
#pragma unroll
  for (int i = 0; i < 4; ++i) f.u[4 + i] = p[8 + hi * 4 + i];
  return f.v;
}

#if USE_TDM
// Issue a TDM 2D tile load: 128 rows x 32 bf16 cols from a [*, 1024] bf16
// row-major tensor into LDS with 17-DWORD row pitch (16 data + 1 pad DWORD).
__device__ __forceinline__ void tdm_load_tile_bf16(const unsigned short* gtile,
                                                   unsigned int lds_off) {
  const unsigned long long ga = (unsigned long long)(size_t)gtile;
  v4u g0;
  g0[0] = 1u;                                     // count=1, user descriptor
  g0[1] = lds_off;                                // lds_addr (bytes)
  g0[2] = (unsigned int)ga;                       // global_addr[31:0]
  g0[3] = (unsigned int)((ga >> 32) & 0x01FFFFFFull) | 0x80000000u;  // addr[56:32] | type=2
  v8i g1;
  g1[0] = (int)((1u << 16)      // data_size = 2 bytes
              | (1u << 20)      // pad_enable
              | (3u << 22)      // pad_interval: 16 DWORDs (= one 64B row)
              | (0u << 25));    // pad_amount: 1 DWORD (-> 68B LDS pitch)
  g1[1] = (int)(1024u << 16);   // tensor_dim0 = 1024 elements/row
  g1[2] = (int)(1024u << 16);   // tensor_dim1 = 1024 rows
  g1[3] = (int)(32u << 16);     // tile_dim0 = 32 (contiguous)
  g1[4] = 128;                  // tile_dim1 = 128 rows, tile_dim2 = 0
  g1[5] = 1024;                 // tensor_dim0_stride[31:0] = 1024 elements
  g1[6] = 0;
  g1[7] = 0;
  v4i z4 = {};                  // groups 2/3 unused for 2D tensors
  v8i z8 = {};
  __builtin_amdgcn_tensor_load_to_lds(g0, g1, z4, z4, z8, 0);
}
#endif

// ---- kernel 1: Q/K/V projections (bf16 WMMA GEMM, fp32 accumulate) ---------
// gridDim = (8, 8, 12); z -> proj = z%3 (0=Q,1=K,2=V), head = z/3
__global__ __launch_bounds__(256) void qkv_proj_kernel(
    const float* __restrict__ node1, const float* __restrict__ node2,
    const float* __restrict__ Wq, const float* __restrict__ bq,
    const float* __restrict__ Wk, const float* __restrict__ bk,
    const float* __restrict__ Wv, const float* __restrict__ bv,
    unsigned short* __restrict__ Qb,   // [h][n][f] bf16
    unsigned short* __restrict__ Kb,   // [h][m][f] bf16 (row-major, same as Q)
    float* __restrict__ Vf)            // [h][m][f] f32
{
  __shared__ unsigned short ldsA[128 * LDS_STRIDE];
  __shared__ unsigned short ldsB[128 * LDS_STRIDE];

  const int z    = blockIdx.z;
  const int proj = z % 3;
  const int h    = z / 3;

  const float* X = (proj == 0) ? node1 : node2;
  const float* W = (proj == 0 ? Wq : (proj == 1 ? Wk : Wv)) + (size_t)h * H_DIM * D_HEAD;
  const float* bias = (proj == 0 ? bq : (proj == 1 ? bk : bv)) + h * D_HEAD;

  const int tid   = threadIdx.x;
  const int lane  = tid & 31;
  const int wave  = tid >> 5;
  const int waveM = wave >> 1;   // 0..3 -> 32 rows each
  const int waveN = wave & 1;    // 0..1 -> 64 cols each
  const int blockM = blockIdx.y * 128;
  const int blockN = blockIdx.x * 128;

  v8f acc[2][4];
#pragma unroll
  for (int i = 0; i < 2; ++i)
#pragma unroll
    for (int j = 0; j < 4; ++j) acc[i][j] = {};

  for (int kt = 0; kt < H_DIM; kt += 32) {
    // A tile: 128 rows x 32 K, fp32 -> bf16
#pragma unroll
    for (int i = 0; i < 4; ++i) {
      const int u = tid + 256 * i;
      const int r = u >> 3, c4 = u & 7;
      const float4 v = *(const float4*)(X + (size_t)(blockM + r) * H_DIM + kt + c4 * 4);
      unsigned int* d = (unsigned int*)(ldsA + r * LDS_STRIDE + c4 * 4);
      d[0] = ((unsigned int)f2bf(v.y) << 16) | f2bf(v.x);
      d[1] = ((unsigned int)f2bf(v.w) << 16) | f2bf(v.z);
    }
    // B tile: 32 K x 128 cols, fp32 -> bf16, stored K-contiguous per column
#pragma unroll
    for (int i = 0; i < 4; ++i) {
      const int u = tid + 256 * i;
      const int k = u >> 5, n4 = u & 31;
      const float4 v = *(const float4*)(W + (size_t)(kt + k) * D_HEAD + blockN + n4 * 4);
      ldsB[(n4 * 4 + 0) * LDS_STRIDE + k] = f2bf(v.x);
      ldsB[(n4 * 4 + 1) * LDS_STRIDE + k] = f2bf(v.y);
      ldsB[(n4 * 4 + 2) * LDS_STRIDE + k] = f2bf(v.z);
      ldsB[(n4 * 4 + 3) * LDS_STRIDE + k] = f2bf(v.w);
    }
    if (kt + 32 < H_DIM) {  // hint next K tile into cache (global_prefetch_b8)
      __builtin_prefetch(X + (size_t)(blockM + (tid >> 1)) * H_DIM + kt + 32, 0, 1);
      __builtin_prefetch(W + (size_t)(kt + 32 + (tid & 31)) * D_HEAD + blockN, 0, 1);
    }
    __syncthreads();

    v16bf afrag[2], bfrag[4];
#pragma unroll
    for (int i = 0; i < 2; ++i) afrag[i] = load_frag_a(ldsA, waveM * 32 + i * 16, lane);
#pragma unroll
    for (int j = 0; j < 4; ++j) bfrag[j] = load_frag_b(ldsB, waveN * 64 + j * 16, lane);
#pragma unroll
    for (int i = 0; i < 2; ++i)
#pragma unroll
      for (int j = 0; j < 4; ++j)
        acc[i][j] = __builtin_amdgcn_wmma_f32_16x16x32_bf16(
            false, afrag[i], false, bfrag[j], (short)0, acc[i][j], false, false);
    __syncthreads();
  }

  // epilogue: bias add + coalesced stores (Q/K bf16 row-major, V f32)
  const int nlo = lane & 15;
  const int rhi = (lane >> 4) * 8;
  const size_t hbase = (size_t)h * N_NODES * D_HEAD;
#pragma unroll
  for (int i = 0; i < 2; ++i) {
#pragma unroll
    for (int j = 0; j < 4; ++j) {
      const int col = blockN + waveN * 64 + j * 16 + nlo;
      const float bval = bias[col];
#pragma unroll
      for (int r = 0; r < 8; ++r) {
        const int row = blockM + waveM * 32 + i * 16 + rhi + r;
        const float val = acc[i][j][r] + bval;
        const size_t idx = hbase + (size_t)row * D_HEAD + col;
        if (proj == 0)       Qb[idx] = f2bf(val);
        else if (proj == 1)  Kb[idx] = f2bf(val);
        else                 Vf[idx] = val;
      }
    }
  }
}

// ---- kernel 2: scores = (Q @ K^T) * scale * V, scattered into d_out --------
// gridDim = (8, 8, 4). TDM double-buffered tile DMA into LDS.
__global__ __launch_bounds__(256) void attn_kernel(
    const unsigned short* __restrict__ Qb, const unsigned short* __restrict__ Kb,
    const float* __restrict__ Vf, float* __restrict__ out)
{
  __shared__ unsigned short ldsA[2][128 * LDS_STRIDE];
  __shared__ unsigned short ldsB[2][128 * LDS_STRIDE];

  const int h = blockIdx.z;
  const size_t hbase = (size_t)h * N_NODES * D_HEAD;
  const unsigned short* Q = Qb + hbase;
  const unsigned short* K = Kb + hbase;
  const float* V = Vf + hbase;

  const int tid   = threadIdx.x;
  const int lane  = tid & 31;
  const int wave  = tid >> 5;
  const int waveM = wave >> 1;
  const int waveN = wave & 1;
  const int blockM = blockIdx.y * 128;
  const int blockN = blockIdx.x * 128;
  const int NT = D_HEAD / 32;   // 32 K-tiles

  v8f acc[2][4];
#pragma unroll
  for (int i = 0; i < 2; ++i)
#pragma unroll
    for (int j = 0; j < 4; ++j) acc[i][j] = {};

#if USE_TDM
  if (wave == 0) {  // prologue: DMA tile 0 into buffer 0
    tdm_load_tile_bf16(Q + (size_t)blockM * D_HEAD,
                       (unsigned int)(size_t)(const void*)&ldsA[0][0]);
    tdm_load_tile_bf16(K + (size_t)blockN * D_HEAD,
                       (unsigned int)(size_t)(const void*)&ldsB[0][0]);
  }
#endif

  for (int t = 0; t < NT; ++t) {
    const int buf = t & 1;
    const int kt  = t * 32;
    (void)kt;
#if USE_TDM
    if (wave == 0) {
      if (t + 1 < NT) {  // DMA next tile into other buffer, then wait for tile t
        tdm_load_tile_bf16(Q + (size_t)blockM * D_HEAD + (t + 1) * 32,
                           (unsigned int)(size_t)(const void*)&ldsA[buf ^ 1][0]);
        tdm_load_tile_bf16(K + (size_t)blockN * D_HEAD + (t + 1) * 32,
                           (unsigned int)(size_t)(const void*)&ldsB[buf ^ 1][0]);
        __builtin_amdgcn_s_wait_tensorcnt(2);
      } else {
        __builtin_amdgcn_s_wait_tensorcnt(0);
      }
    }
#else
    // fallback: VGPR staging (bf16 tiles are row-major in both Q and K)
#pragma unroll
    for (int i = 0; i < 4; ++i) {
      const int u = tid + 256 * i;
      const int r = u >> 3, c4 = u & 7;
      const uint2 va = *(const uint2*)(Q + (size_t)(blockM + r) * D_HEAD + kt + c4 * 4);
      unsigned int* da = (unsigned int*)(&ldsA[buf][0] + r * LDS_STRIDE + c4 * 4);
      da[0] = va.x; da[1] = va.y;
      const uint2 vb = *(const uint2*)(K + (size_t)(blockN + r) * D_HEAD + kt + c4 * 4);
      unsigned int* db = (unsigned int*)(&ldsB[buf][0] + r * LDS_STRIDE + c4 * 4);
      db[0] = vb.x; db[1] = vb.y;
    }
#endif
    __syncthreads();

    v16bf afrag[2], bfrag[4];
#pragma unroll
    for (int i = 0; i < 2; ++i) afrag[i] = load_frag_a(&ldsA[buf][0], waveM * 32 + i * 16, lane);
#pragma unroll
    for (int j = 0; j < 4; ++j) bfrag[j] = load_frag_b(&ldsB[buf][0], waveN * 64 + j * 16, lane);
#pragma unroll
    for (int i = 0; i < 2; ++i)
#pragma unroll
      for (int j = 0; j < 4; ++j)
        acc[i][j] = __builtin_amdgcn_wmma_f32_16x16x32_bf16(
            false, afrag[i], false, bfrag[j], (short)0, acc[i][j], false, false);
    __syncthreads();  // all waves done reading buf before TDM overwrites it
  }

  // epilogue: r[h,n,m] = score*scale*V[n,m]; out[n, h*1024+m] (pre-LN)
  const float scale = 0.03125f;  // 1/sqrt(1024)
  const int nlo = lane & 15;
  const int rhi = (lane >> 4) * 8;
#pragma unroll
  for (int i = 0; i < 2; ++i) {
#pragma unroll
    for (int j = 0; j < 4; ++j) {
      const int col = blockN + waveN * 64 + j * 16 + nlo;
#pragma unroll
      for (int r = 0; r < 8; ++r) {
        const int row = blockM + waveM * 32 + i * 16 + rhi + r;
        const float val = acc[i][j][r] * scale * V[(size_t)row * D_HEAD + col];
        out[(size_t)row * H_DIM + (size_t)h * D_HEAD + col] = val;
      }
    }
  }
}

// ---- kernel 3: in-place LayerNorm over rows of d_out -----------------------
__global__ __launch_bounds__(256) void ln_kernel(
    float* __restrict__ out, const float* __restrict__ gamma, const float* __restrict__ beta)
{
  const int row  = blockIdx.x;
  const int tid  = threadIdx.x;
  const int lane = tid & 31;
  const int wave = tid >> 5;
  float* rp = out + (size_t)row * H_DIM;

  float4 x[4];
  float s = 0.f, s2 = 0.f;
#pragma unroll
  for (int i = 0; i < 4; ++i) {
    x[i] = *(const float4*)(rp + i * 1024 + tid * 4);
    s  += x[i].x + x[i].y + x[i].z + x[i].w;
    s2 += x[i].x * x[i].x + x[i].y * x[i].y + x[i].z * x[i].z + x[i].w * x[i].w;
  }
#pragma unroll
  for (int off = 16; off > 0; off >>= 1) {
    s  += __shfl_xor(s, off, 32);
    s2 += __shfl_xor(s2, off, 32);
  }
  __shared__ float red[16];
  if (lane == 0) { red[wave] = s; red[8 + wave] = s2; }
  __syncthreads();
  s = 0.f; s2 = 0.f;
#pragma unroll
  for (int w = 0; w < 8; ++w) { s += red[w]; s2 += red[8 + w]; }
  const float inv = 1.0f / (float)H_DIM;
  const float mu  = s * inv;
  const float var = s2 * inv - mu * mu;
  const float rs  = rsqrtf(var + LN_EPS);
#pragma unroll
  for (int i = 0; i < 4; ++i) {
    const int c = i * 1024 + tid * 4;
    const float4 g = *(const float4*)(gamma + c);
    const float4 b = *(const float4*)(beta + c);
    float4 y;
    y.x = (x[i].x - mu) * rs * g.x + b.x;
    y.y = (x[i].y - mu) * rs * g.y + b.y;
    y.z = (x[i].z - mu) * rs * g.z + b.z;
    y.w = (x[i].w - mu) * rs * g.w + b.w;
    *(float4*)(rp + c) = y;
  }
}

// ---- launcher --------------------------------------------------------------

extern "C" void kernel_launch(void* const* d_in, const int* in_sizes, int n_in,
                              void* d_out, int out_size, void* d_ws, size_t ws_size,
                              hipStream_t stream) {
  const float* node1 = (const float*)d_in[0];
  const float* node2 = (const float*)d_in[1];
  const float* Wq    = (const float*)d_in[2];
  const float* bq    = (const float*)d_in[3];
  const float* Wk    = (const float*)d_in[4];
  const float* bk    = (const float*)d_in[5];
  const float* Wv    = (const float*)d_in[6];
  const float* bv    = (const float*)d_in[7];
  const float* gamma = (const float*)d_in[8];
  const float* beta  = (const float*)d_in[9];
  float* out = (float*)d_out;

  const size_t headElems = (size_t)N_HEADS * N_NODES * D_HEAD;  // 4Mi
  unsigned short* Qb = (unsigned short*)d_ws;                   // 8 MB bf16
  unsigned short* Kb = Qb + headElems;                          // 8 MB bf16
  float* Vf = (float*)(Kb + headElems);                         // 16 MB f32

  dim3 blk(256);
  dim3 g1(8, 8, 12);
  qkv_proj_kernel<<<g1, blk, 0, stream>>>(node1, node2, Wq, bq, Wk, bk, Wv, bv, Qb, Kb, Vf);
  dim3 g2(8, 8, 4);
  attn_kernel<<<g2, blk, 0, stream>>>(Qb, Kb, Vf, out);
  ln_kernel<<<dim3(N_NODES), blk, 0, stream>>>(out, gamma, beta);
}
